// LSTM_24627342475594
// MI455X (gfx1250) — matrix operand order repaired and checked by reading, not compiled, not run on hardware
//
#include <hip/hip_runtime.h>

typedef __attribute__((ext_vector_type(16))) _Float16 v16h;
typedef __attribute__((ext_vector_type(8)))  float    v8f;

#define B_TOT 4096
#define T_LEN 2048
#define I_DIM 4
#define H_DIM 20
#define O_DIM 4
#define WAVES_PER_BLK 4
#define ROWS_PER_WAVE 16

// Prefer the CDNA5 native transcendental tanh (single TRANS op) if the
// toolchain declares it; otherwise use exp2/rcp with the 2*log2(e) factor
// pre-folded into the weight matrix (input arrives pre-scaled).
#if __has_builtin(__builtin_amdgcn_tanhf)
  #define HAVE_NATIVE_TANH 1
  __device__ __forceinline__ float fast_tanh(float v) {
    return __builtin_amdgcn_tanhf(v);
  }
#elif __has_builtin(__builtin_amdgcn_tanh_f32)
  #define HAVE_NATIVE_TANH 1
  __device__ __forceinline__ float fast_tanh(float v) {
    return __builtin_amdgcn_tanh_f32(v);
  }
#else
  #define HAVE_NATIVE_TANH 0
  // v is already scaled by 2*log2(e): exp2(v) = e^{2x}
  // tanh(x) = 1 - 2/(e^{2x}+1); saturates correctly at +-1.
  __device__ __forceinline__ float fast_tanh(float v) {
    float e = __builtin_amdgcn_exp2f(v);
    float r = __builtin_amdgcn_rcpf(e + 1.0f);
    return __builtin_fmaf(-2.0f, r, 1.0f);
  }
#endif

// Weight pre-scale: identity for native tanh, 2*log2(e) for exp2 fallback.
#if HAVE_NATIVE_TANH
  #define WSCALE 1.0f
#else
  #define WSCALE 2.88539008177792681472f
#endif

// Swap lane halves (lane i <-> lane i^16) in one VALU op, no LDS.
__device__ __forceinline__ float swap16(float v) {
  int i = __builtin_bit_cast(int, v);
  int r = __builtin_amdgcn_permlanex16(i, i, 0x76543210, 0xfedcba98, false, false);
  return __builtin_bit_cast(float, r);
}

// Fused tanh-RNN scan, register-resident state:
//   A (constant) = [W_hh | W_ih | b]^T padded to 32x32, two 16x32 f16 tiles
//   B (state)    = [h | x_t | 1 | 0]   (K=32 x 16 batch cols, f16)
//   D = A x B -> tanh -> permlanex16 half-swap -> next B. No LDS in the loop.
__global__ __launch_bounds__(WAVES_PER_BLK * 32, 1)
void rnn_wmma_scan_kernel(const float* __restrict__ x,
                          const float* __restrict__ h0,
                          const float* __restrict__ W_ih,
                          const float* __restrict__ W_hh,
                          const float* __restrict__ b_ih,
                          const float* __restrict__ b_hh,
                          const float* __restrict__ fc_w,
                          const float* __restrict__ fc_b,
                          float* __restrict__ out) {
  __shared__ float s_mt[32][32];  // augmented weight matrix M^T: [K][j], zero-padded
  __shared__ float s_fcw[O_DIM * H_DIM];
  __shared__ float s_fcb[O_DIM];

  const int tid  = threadIdx.x;
  const int wid  = tid >> 5;
  const int lane = tid & 31;
  const int b0   = blockIdx.x * (WAVES_PER_BLK * ROWS_PER_WAVE) + wid * ROWS_PER_WAVE;

  // ---- build padded augmented weight matrix in LDS (block-cooperative) ----
  // Pre-scaled by WSCALE so the tanh stage needs no multiply (fallback path).
  for (int idx = tid; idx < 32 * 32; idx += blockDim.x) {
    const int K = idx >> 5, j = idx & 31;
    float v = 0.0f;
    if (j < H_DIM) {
      if (K < H_DIM)               v = W_hh[j * H_DIM + K];            // h -> h
      else if (K < H_DIM + I_DIM)  v = W_ih[j * I_DIM + (K - H_DIM)];  // x -> h
      else if (K == H_DIM + I_DIM) v = b_ih[j] + b_hh[j];              // bias lane
    }
    s_mt[K][j] = v * WSCALE;
  }
  for (int i = tid; i < O_DIM * H_DIM; i += blockDim.x) s_fcw[i] = fc_w[i];
  for (int i = tid; i < O_DIM; i += blockDim.x)         s_fcb[i] = fc_b[i];
  __syncthreads();

  const int  col = lane & 15;   // batch column within tile / weight output row
  const bool lo  = (lane < 16);

  // ---- constant A operands (weights) in WMMA A layout ----
  // A (f16 16x32): lanes 0-15 row M=lane:     e -> K=e      (0..15)
  //                lanes 16-31 row M=lane-16: e -> K=16+e   (16..31)
  v16h a0w, a1w;
#pragma unroll
  for (int e = 0; e < 16; ++e) {
    const int K = lo ? e : e + 16;
    a0w[e] = (_Float16)s_mt[K][col];        // output rows j = 0..15
    a1w[e] = (_Float16)s_mt[K][16 + col];   // output rows j = 16..31 (16..19 useful)
  }

  // ---- x stream: 16B per (row, t); only lanes 16-31 consume it ----
  const float4* __restrict__ xrow =
      (const float4*)(x + (size_t)(b0 + col) * T_LEN * I_DIM);
  float4 xt = xrow[0];

  // ---- initial B (state) operand from h0 + x_0 ----
  // B (f16 32x16): lanes 0-15 col N=lane:     e -> K=e      (h[0..15])
  //                lanes 16-31 col N=lane-16: e -> K=16+e   (h16..19 | x | 1 | 0)
  const float* h0row = h0 + (size_t)(b0 + col) * H_DIM;
  v16h bst;
  if (lo) {
#pragma unroll
    for (int e = 0; e < 16; ++e) bst[e] = (_Float16)h0row[e];
  } else {
#pragma unroll
    for (int e = 0; e < 4; ++e) bst[e] = (_Float16)h0row[16 + e];
    bst[4] = (_Float16)xt.x; bst[5] = (_Float16)xt.y;
    bst[6] = (_Float16)xt.z; bst[7] = (_Float16)xt.w;
    bst[8] = (_Float16)1.0f;
#pragma unroll
    for (int e = 9; e < 16; ++e) bst[e] = (_Float16)0.0f;
  }

  // ---- scan over T: state never leaves registers ----
  float t0[8], s0[8], t1[4], s1[4];
  for (int t = 0; t < T_LEN; ++t) {
    float4 xn = (t + 1 < T_LEN) ? xrow[t + 1] : make_float4(0.f, 0.f, 0.f, 0.f);
    if ((t & 7) == 0) __builtin_prefetch(&xrow[t + 16], 0, 3);  // one per 128B line

    v8f c = {};
    v8f d0 = __builtin_amdgcn_wmma_f32_16x16x32_f16(false, a0w, false, bst,
                                                    (short)0, c, false, false);
    v8f d1 = __builtin_amdgcn_wmma_f32_16x16x32_f16(false, a1w, false, bst,
                                                    (short)0, c, false, false);

    // h_{t+1}: lanes 0-15 of d0 hold h[0..7], lanes 16-31 hold h[8..15];
    //          lanes 0-15 of d1 hold h[16..19] (rows 20..31 are zero-weight)
#pragma unroll
    for (int r = 0; r < 8; ++r) t0[r] = fast_tanh(d0[r]);
#pragma unroll
    for (int r = 0; r < 4; ++r) t1[r] = fast_tanh(d1[r]);
#pragma unroll
    for (int r = 0; r < 8; ++r) s0[r] = swap16(t0[r]);
#pragma unroll
    for (int r = 0; r < 4; ++r) s1[r] = swap16(t1[r]);

    // assemble next B operand (x_{t+1} enters at K=20..23 -> hi-half e=4..7)
    bst[0]  = (_Float16)(lo ? t0[0] : s1[0]);
    bst[1]  = (_Float16)(lo ? t0[1] : s1[1]);
    bst[2]  = (_Float16)(lo ? t0[2] : s1[2]);
    bst[3]  = (_Float16)(lo ? t0[3] : s1[3]);
    bst[4]  = (_Float16)(lo ? t0[4] : xn.x);
    bst[5]  = (_Float16)(lo ? t0[5] : xn.y);
    bst[6]  = (_Float16)(lo ? t0[6] : xn.z);
    bst[7]  = (_Float16)(lo ? t0[7] : xn.w);
    bst[8]  = (_Float16)(lo ? s0[0] : 1.0f);
    bst[9]  = (_Float16)(lo ? s0[1] : 0.0f);
    bst[10] = (_Float16)(lo ? s0[2] : 0.0f);
    bst[11] = (_Float16)(lo ? s0[3] : 0.0f);
    bst[12] = (_Float16)(lo ? s0[4] : 0.0f);
    bst[13] = (_Float16)(lo ? s0[5] : 0.0f);
    bst[14] = (_Float16)(lo ? s0[6] : 0.0f);
    bst[15] = (_Float16)(lo ? s0[7] : 0.0f);
    xt = xn;
  }

  // ---- final h per lane (both halves hold the full 20-vector) ----
  // lanes 0-15:  h[0..7]=t0, h[8..15]=s0, h[16..19]=t1
  // lanes 16-31: h[0..7]=s0, h[8..15]=t0, h[16..19]=s1
  float h[H_DIM];
#pragma unroll
  for (int r = 0; r < 8; ++r) h[r] = lo ? t0[r] : s0[r];
#pragma unroll
  for (int r = 0; r < 8; ++r) h[8 + r] = lo ? s0[r] : t0[r];
#pragma unroll
  for (int r = 0; r < 4; ++r) h[16 + r] = lo ? t1[r] : s1[r];

  // ---- final FC: each lane computes 2 of the 64 outputs of its wave tile ----
  const int ob = (lane >> 4) * 2;  // lanes 0-15 -> o in {0,1}; lanes 16-31 -> {2,3}
  float acc0 = s_fcb[ob], acc1 = s_fcb[ob + 1];
#pragma unroll
  for (int k = 0; k < H_DIM; ++k) {
    acc0 = __builtin_fmaf(h[k], s_fcw[ob * H_DIM + k], acc0);
    acc1 = __builtin_fmaf(h[k], s_fcw[(ob + 1) * H_DIM + k], acc1);
  }
  out[(size_t)(b0 + col) * O_DIM + ob]     = acc0;
  out[(size_t)(b0 + col) * O_DIM + ob + 1] = acc1;
}

extern "C" void kernel_launch(void* const* d_in, const int* in_sizes, int n_in,
                              void* d_out, int out_size, void* d_ws, size_t ws_size,
                              hipStream_t stream) {
  (void)in_sizes; (void)n_in; (void)out_size; (void)d_ws; (void)ws_size;
  const float* x    = (const float*)d_in[0];
  const float* h0   = (const float*)d_in[1];
  const float* W_ih = (const float*)d_in[2];
  const float* W_hh = (const float*)d_in[3];
  const float* b_ih = (const float*)d_in[4];
  const float* b_hh = (const float*)d_in[5];
  const float* fc_w = (const float*)d_in[6];
  const float* fc_b = (const float*)d_in[7];
  float* out = (float*)d_out;

  const int rows_per_block = WAVES_PER_BLK * ROWS_PER_WAVE;  // 64
  dim3 grid(B_TOT / rows_per_block);                         // 64 blocks
  dim3 block(WAVES_PER_BLK * 32);                            // 128 threads = 4 waves
  rnn_wmma_scan_kernel<<<grid, block, 0, stream>>>(x, h0, W_ih, W_hh, b_ih, b_hh,
                                                   fc_w, fc_b, out);
}
